// MM_GCN_54735063220453
// MI455X (gfx1250) — matrix operand order repaired and checked by reading, not compiled, not run on hardware
//
#include <hip/hip_runtime.h>
#include <hip/hip_bf16.h>

// ---------------------------------------------------------------------------
// MM-GCN on gfx1250 (MI455X).  Dominant matmul adj@h done with
// v_wmma_f32_16x16x32_bf16 (f32 accumulate).  adj converted once to bf16
// (72 MB -> L2 resident across all 8 layers).
// ---------------------------------------------------------------------------

typedef __bf16 bf16_t;
typedef bf16_t v16bf  __attribute__((ext_vector_type(16)));
typedef bf16_t bf16x8 __attribute__((ext_vector_type(8)));
typedef float  v8f    __attribute__((ext_vector_type(8)));

union V16 { v16bf v; bf16x8 h[2]; };

// ---- problem constants (match reference) ----------------------------------
#define NROW      2000      // utterances per modality
#define NODES     6000      // 3 * NROW graph nodes
#define DIM       200       // n_dim == nhidden
#define NLAYERS   8
#define SEQL      100
#define NDLG      20

// ---- padded layout constants ----------------------------------------------
#define KA        320       // A_DIM 300 -> pad32
#define KV        352       // V_DIM 342 -> pad32
#define KL        1024
#define LD_X      224       // DIM 200 -> pad32 (K of h0 gemm)
#define LD_SUP    416       // 2*DIM 400 -> pad32
#define LD_ADJ    6016      // 6000 -> pad32 (also pad128 for A rows)
#define ROWS_PROJ 2048      // 2000 -> pad128
#define ROWS_ND   6016      // 6000 -> pad128
#define NB_ROWS   224       // B-operand row pad (N=200 -> grid covers 224)

// ===========================================================================
// Wave-level 32x32 GEMM core:  C[M x N] = A[M x K] * B[N x K]^T  (bf16 in,
// f32 accum).  Each wave owns a 32(M) x 32(N) tile; 4 WMMA per 32-wide K step.
// VGPR layouts per CDNA5 ISA 7.12.2:
//   A lane(l): row = m0 + (l&15) [+16 for second M tile],
//              elems 0..7  = K k0+8*(l>>4)    .. +7
//              elems 8..15 = K k0+16+8*(l>>4) .. +7
//   B lane(l): col = n0 + (l&15), elems same K order as A
// ===========================================================================
__device__ __forceinline__ void wmma_tile_mm(const bf16_t* __restrict__ A,
                                             const bf16_t* __restrict__ B,
                                             int ldA, int ldB, int K,
                                             int m0, int n0, v8f acc[2][2]) {
  const int lane = threadIdx.x & 31;
  const int hf   = lane >> 4;
  const int lm   = lane & 15;
  const bf16_t* __restrict__ ar0 = A + (size_t)(m0 + lm)      * ldA;
  const bf16_t* __restrict__ ar1 = A + (size_t)(m0 + 16 + lm) * ldA;
  const bf16_t* __restrict__ br0 = B + (size_t)(n0 + lm)      * ldB;
  const bf16_t* __restrict__ br1 = B + (size_t)(n0 + 16 + lm) * ldB;
  const int ao = hf * 8;    // A half-lane K offset
  const int bo = hf * 8;    // B half-lane K offset
  for (int k0 = 0; k0 < K; k0 += 32) {
    V16 a0, a1, b0, b1;
    a0.h[0] = *(const bf16x8*)(ar0 + k0 + ao);
    a0.h[1] = *(const bf16x8*)(ar0 + k0 + 16 + ao);
    a1.h[0] = *(const bf16x8*)(ar1 + k0 + ao);
    a1.h[1] = *(const bf16x8*)(ar1 + k0 + 16 + ao);
    b0.h[0] = *(const bf16x8*)(br0 + k0 + bo);
    b0.h[1] = *(const bf16x8*)(br0 + k0 + 16 + bo);
    b1.h[0] = *(const bf16x8*)(br1 + k0 + bo);
    b1.h[1] = *(const bf16x8*)(br1 + k0 + 16 + bo);
    acc[0][0] = __builtin_amdgcn_wmma_f32_16x16x32_bf16(false, a0.v, false, b0.v, (short)0, acc[0][0], false, false);
    acc[0][1] = __builtin_amdgcn_wmma_f32_16x16x32_bf16(false, a0.v, false, b1.v, (short)0, acc[0][1], false, false);
    acc[1][0] = __builtin_amdgcn_wmma_f32_16x16x32_bf16(false, a1.v, false, b0.v, (short)0, acc[1][0], false, false);
    acc[1][1] = __builtin_amdgcn_wmma_f32_16x16x32_bf16(false, a1.v, false, b1.v, (short)0, acc[1][1], false, false);
  }
}

#define GEMM_PROLOGUE()                                        \
  const int m0 = (blockIdx.x * 4 + (threadIdx.x >> 5)) * 32;   \
  const int n0 = blockIdx.y * 32;                              \
  v8f acc[2][2] = {};

#define EPILOGUE_LOOP(MROWS)                                   \
  const int lane = threadIdx.x & 31;                           \
  const int hf   = lane >> 4;                                  \
  const int lm   = lane & 15;                                  \
  for (int mi = 0; mi < 2; ++mi)                               \
    for (int ni = 0; ni < 2; ++ni)                             \
      for (int e = 0; e < 8; ++e) {                            \
        const int row = m0 + mi * 16 + hf * 8 + e;             \
        const int col = n0 + ni * 16 + lm;                     \
        if (row < (MROWS) && col < DIM)

#define EPILOGUE_END }

// ---- modality projection:  x = relu-free (A @ W^T + b [+ spk_emb]) --------
__global__ void gemm_proj_kernel(const bf16_t* __restrict__ A,
                                 const bf16_t* __restrict__ B,
                                 const float* __restrict__ bias,
                                 const float* __restrict__ spk_emb,
                                 const int* __restrict__ spk_idx, int use_spk,
                                 int ldA, int ldB, int K,
                                 float* __restrict__ x_f32,
                                 bf16_t* __restrict__ xB, int baseRow) {
  GEMM_PROLOGUE();
  wmma_tile_mm(A, B, ldA, ldB, K, m0, n0, acc);
  EPILOGUE_LOOP(NROW) {
    float val = acc[mi][ni][e] + bias[col];
    if (use_spk) val += spk_emb[spk_idx[row] * DIM + col];
    x_f32[(size_t)(baseRow + row) * DIM + col] = val;
    xB[(size_t)(baseRow + row) * LD_X + col]   = (bf16_t)val;
  } EPILOGUE_END
}

// ---- h0 = relu(x @ W0^T + b0); also fill support[:,200:400] and hT --------
__global__ void gemm_h0_kernel(const bf16_t* __restrict__ xB,
                               const bf16_t* __restrict__ W0T,
                               const float* __restrict__ b0,
                               float* __restrict__ h0_f32,
                               bf16_t* __restrict__ supB,
                               bf16_t* __restrict__ hT) {
  GEMM_PROLOGUE();
  wmma_tile_mm(xB, W0T, LD_X, LD_X, LD_X, m0, n0, acc);
  EPILOGUE_LOOP(NODES) {
    float val = fmaxf(acc[mi][ni][e] + b0[col], 0.0f);
    h0_f32[(size_t)row * DIM + col]            = val;
    supB[(size_t)row * LD_SUP + DIM + col]     = (bf16_t)val;   // h0 half (static)
    hT[(size_t)col * LD_ADJ + row]             = (bf16_t)val;   // h := h0 initially
  } EPILOGUE_END
}

// ---- hi = adj @ h  (the big one); also fill support[:,0:200] --------------
__global__ void gemm_adj_kernel(const bf16_t* __restrict__ adjB,
                                const bf16_t* __restrict__ hT,
                                float* __restrict__ hi_f32,
                                bf16_t* __restrict__ supB) {
  GEMM_PROLOGUE();
  wmma_tile_mm(adjB, hT, LD_ADJ, LD_ADJ, LD_ADJ, m0, n0, acc);
  EPILOGUE_LOOP(NODES) {
    float val = acc[mi][ni][e];
    hi_f32[(size_t)row * DIM + col]  = val;
    supB[(size_t)row * LD_SUP + col] = (bf16_t)val;
  } EPILOGUE_END
}

// ---- h = relu(theta*(support @ W^T) + (1-theta)*(0.9*hi + 0.1*h0)) --------
__global__ void gemm_layer_kernel(const bf16_t* __restrict__ supB,
                                  const bf16_t* __restrict__ cwT,
                                  const float* __restrict__ hi_f32,
                                  const float* __restrict__ h0_f32,
                                  float theta,
                                  float* __restrict__ h_f32,
                                  bf16_t* __restrict__ hT) {
  GEMM_PROLOGUE();
  wmma_tile_mm(supB, cwT, LD_SUP, LD_SUP, LD_SUP, m0, n0, acc);
  EPILOGUE_LOOP(NODES) {
    const size_t o = (size_t)row * DIM + col;
    float r   = 0.9f * hi_f32[o] + 0.1f * h0_f32[o];
    float val = fmaxf(theta * acc[mi][ni][e] + (1.0f - theta) * r, 0.0f);
    h_f32[o]                       = val;
    hT[(size_t)col * LD_ADJ + row] = (bf16_t)val;
  } EPILOGUE_END
}

// ---- converters ------------------------------------------------------------
__global__ void cvt_kernel(const float* __restrict__ src, bf16_t* __restrict__ dst,
                           int R, int C, int ldD) {
  int idx = blockIdx.x * blockDim.x + threadIdx.x;
  if (idx >= R * C) return;
  int r = idx / C, c = idx % C;
  dst[(size_t)r * ldD + c] = (bf16_t)src[idx];
}

__global__ void cvtT_kernel(const float* __restrict__ src, bf16_t* __restrict__ dst,
                            int K, int Nn, int ldD) {
  int idx = blockIdx.x * blockDim.x + threadIdx.x;
  if (idx >= K * Nn) return;
  int k = idx / Nn, n = idx % Nn;
  dst[(size_t)n * ldD + k] = (bf16_t)src[idx];
}

__global__ void cvtT_conv_kernel(const float* __restrict__ src, bf16_t* __restrict__ dst) {
  int idx = blockIdx.x * blockDim.x + threadIdx.x;
  if (idx >= NLAYERS * 2 * DIM * DIM) return;
  int t   = idx / (2 * DIM * DIM);
  int rem = idx % (2 * DIM * DIM);
  int k = rem / DIM, n = rem % DIM;
  dst[(size_t)t * NB_ROWS * LD_SUP + (size_t)n * LD_SUP + k] = (bf16_t)src[idx];
}

// ---- speaker argmax (NSPK == 2) -------------------------------------------
__global__ void spk_kernel(const float* __restrict__ qmask, int* __restrict__ spk_idx) {
  int i = blockIdx.x * blockDim.x + threadIdx.x;
  if (i >= NROW) return;
  int b = i / SEQL, t = i % SEQL;
  float q0 = qmask[((size_t)t * NDLG + b) * 2 + 0];
  float q1 = qmask[((size_t)t * NDLG + b) * 2 + 1];
  spk_idx[i] = (q0 >= q1) ? 0 : 1;   // argmax, first index wins ties
}

// ---- output gather: [2000 x 1200] = per-modality [x | h] fused ------------
__global__ void out_kernel(const float* __restrict__ x, const float* __restrict__ h,
                           float* __restrict__ out) {
  int idx = blockIdx.x * blockDim.x + threadIdx.x;
  if (idx >= NROW * 1200) return;
  int i  = idx / 1200;
  int j  = idx % 1200;
  int c  = j / 400;
  int jj = j % 400;
  size_t r = (size_t)(c * NROW + i);
  out[idx] = (jj < DIM) ? x[r * DIM + jj] : h[r * DIM + (jj - DIM)];
}

// ===========================================================================
extern "C" void kernel_launch(void* const* d_in, const int* in_sizes, int n_in,
                              void* d_out, int out_size, void* d_ws, size_t ws_size,
                              hipStream_t stream) {
  (void)in_sizes; (void)n_in; (void)out_size; (void)ws_size;
  const float* a       = (const float*)d_in[0];
  const float* v       = (const float*)d_in[1];
  const float* l       = (const float*)d_in[2];
  const float* qmask   = (const float*)d_in[3];
  const float* adj     = (const float*)d_in[4];
  const float* Wa      = (const float*)d_in[5];
  const float* ba      = (const float*)d_in[6];
  const float* Wv      = (const float*)d_in[7];
  const float* bv      = (const float*)d_in[8];
  const float* Wl      = (const float*)d_in[9];
  const float* bl      = (const float*)d_in[10];
  const float* spk_emb = (const float*)d_in[11];
  const float* W0      = (const float*)d_in[12];
  const float* b0      = (const float*)d_in[13];
  const float* convW   = (const float*)d_in[14];
  float* out = (float*)d_out;

  // ---- workspace carve-out (bf16 region first -> single memset) -----------
  char* ws = (char*)d_ws;
  size_t off = 0;
  auto take = [&](size_t bytes) -> void* {
    off = (off + 255) & ~(size_t)255;
    void* p = ws + off;
    off += bytes;
    return p;
  };
  bf16_t* adjB  = (bf16_t*)take((size_t)ROWS_ND * LD_ADJ * 2);      // 72.4 MB
  bf16_t* hT    = (bf16_t*)take((size_t)NB_ROWS * LD_ADJ * 2);
  bf16_t* xB    = (bf16_t*)take((size_t)ROWS_ND * LD_X * 2);
  bf16_t* supB  = (bf16_t*)take((size_t)ROWS_ND * LD_SUP * 2);
  bf16_t* aB    = (bf16_t*)take((size_t)ROWS_PROJ * KA * 2);
  bf16_t* vB    = (bf16_t*)take((size_t)ROWS_PROJ * KV * 2);
  bf16_t* lB    = (bf16_t*)take((size_t)ROWS_PROJ * KL * 2);
  bf16_t* WaT   = (bf16_t*)take((size_t)NB_ROWS * KA * 2);
  bf16_t* WvT   = (bf16_t*)take((size_t)NB_ROWS * KV * 2);
  bf16_t* WlT   = (bf16_t*)take((size_t)NB_ROWS * KL * 2);
  bf16_t* W0T   = (bf16_t*)take((size_t)NB_ROWS * LD_X * 2);
  bf16_t* cwT   = (bf16_t*)take((size_t)NLAYERS * NB_ROWS * LD_SUP * 2);
  size_t bf16_end = (off + 255) & ~(size_t)255;
  float* x_f32  = (float*)take((size_t)NODES * DIM * 4);
  float* h0_f32 = (float*)take((size_t)NODES * DIM * 4);
  float* hi_f32 = (float*)take((size_t)NODES * DIM * 4);
  float* h_f32  = (float*)take((size_t)NODES * DIM * 4);
  int*   spkI   = (int*)take((size_t)NROW * 4);

  // zero all bf16 pads (harness poisons ws with 0xAA)
  hipMemsetAsync(d_ws, 0, bf16_end, stream);

  // ---- prep: speaker ids + bf16 conversions -------------------------------
  const int CT = 256;
  spk_kernel<<<(NROW + CT - 1) / CT, CT, 0, stream>>>(qmask, spkI);
  cvt_kernel<<<((size_t)NODES * NODES + CT - 1) / CT, CT, 0, stream>>>(adj, adjB, NODES, NODES, LD_ADJ);
  cvt_kernel<<<(NROW * 300 + CT - 1) / CT, CT, 0, stream>>>(a, aB, NROW, 300, KA);
  cvt_kernel<<<(NROW * 342 + CT - 1) / CT, CT, 0, stream>>>(v, vB, NROW, 342, KV);
  cvt_kernel<<<(NROW * 1024 + CT - 1) / CT, CT, 0, stream>>>(l, lB, NROW, 1024, KL);
  cvtT_kernel<<<(300 * DIM + CT - 1) / CT, CT, 0, stream>>>(Wa, WaT, 300, DIM, KA);
  cvtT_kernel<<<(342 * DIM + CT - 1) / CT, CT, 0, stream>>>(Wv, WvT, 342, DIM, KV);
  cvtT_kernel<<<(1024 * DIM + CT - 1) / CT, CT, 0, stream>>>(Wl, WlT, 1024, DIM, KL);
  cvtT_kernel<<<(DIM * DIM + CT - 1) / CT, CT, 0, stream>>>(W0, W0T, DIM, DIM, LD_X);
  cvtT_conv_kernel<<<(NLAYERS * 2 * DIM * DIM + CT - 1) / CT, CT, 0, stream>>>(convW, cwT);

  // ---- GEMM geometry: 128-thread blocks = 4 waves of 32x32 tiles ----------
  const dim3 blk(128);
  const dim3 gProj(ROWS_PROJ / 128, NB_ROWS / 32);   // (16, 7)
  const dim3 gNode(ROWS_ND / 128, NB_ROWS / 32);     // (47, 7)

  // modality projections -> x (f32) and xB (bf16)
  gemm_proj_kernel<<<gProj, blk, 0, stream>>>(aB, WaT, ba, spk_emb, spkI, 0,
                                              KA, KA, KA, x_f32, xB, 0);
  gemm_proj_kernel<<<gProj, blk, 0, stream>>>(vB, WvT, bv, spk_emb, spkI, 0,
                                              KV, KV, KV, x_f32, xB, NROW);
  gemm_proj_kernel<<<gProj, blk, 0, stream>>>(lB, WlT, bl, spk_emb, spkI, 1,
                                              KL, KL, KL, x_f32, xB, 2 * NROW);

  // h0 = relu(x @ W0 + b0); also seeds support[:,200:400] and hT (h := h0)
  gemm_h0_kernel<<<gNode, blk, 0, stream>>>(xB, W0T, b0, h0_f32, supB, hT);

  // GCNII layers
  for (int t = 0; t < NLAYERS; ++t) {
    float theta = logf(0.5f / (float)(t + 1) + 1.0f);
    gemm_adj_kernel<<<gNode, blk, 0, stream>>>(adjB, hT, hi_f32, supB);
    gemm_layer_kernel<<<gNode, blk, 0, stream>>>(supB, cwT + (size_t)t * NB_ROWS * LD_SUP,
                                                 hi_f32, h0_f32, theta, h_f32, hT);
  }

  // fuse [x | h] per modality -> [2000 x 1200]
  out_kernel<<<(NROW * 1200 + CT - 1) / CT, CT, 0, stream>>>(x_f32, h_f32, out);
}
